// RWKVEncoder_78580721647784
// MI455X (gfx1250) — matrix operand order repaired and checked by reading, not compiled, not run on hardware
//
#include <hip/hip_runtime.h>

// ---------------------------------------------------------------------------
// RWKV-v4 layer for MI455X (gfx1250, wave32).
//  - GEMMs: v_wmma_f32_16x16x32_bf16, 32x64 wave tiles, 128x128 block tiles,
//    double-buffered LDS staging of A and B; K-loop fully unrolled so LDS
//    offsets are immediates and DS waits batch across iterations.
//  - WKV scan: chunk-parallel 3-phase linear-recurrence scan (16 chunks),
//    exact reproduction of the reference's stabilized (aa,bb,pp) rep.
// ---------------------------------------------------------------------------

#define BB   8
#define TT   1024
#define CC   1024      // HIDDEN
#define BT   (BB*TT)   // 8192 rows for all GEMMs
#define NCH  16        // scan chunks
#define CL   (TT/NCH)  // 64 steps per chunk

typedef __attribute__((ext_vector_type(16))) __bf16 v16bf;
typedef __attribute__((ext_vector_type(8)))  __bf16 v8bf;
typedef __attribute__((ext_vector_type(8)))  float  v8f;

__device__ __forceinline__ __bf16 to_bf16(float f) {
    union { float f; unsigned u; } in; in.f = f;
    unsigned r = in.u + 0x7FFFu + ((in.u >> 16) & 1u);   // round-to-nearest-even
    union { unsigned short u; __bf16 h; } out;
    out.u = (unsigned short)(r >> 16);
    return out.h;
}

// --------------------------- weight transpose ------------------------------
__global__ __launch_bounds__(256) void w_transpose_bf16(
    const float* __restrict__ W, __bf16* __restrict__ Wt) {
    __shared__ float tile[32][33];
    const int tx = threadIdx.x, ty = threadIdx.y;      // 32 x 8
    #pragma unroll
    for (int i = 0; i < 4; ++i) {
        int d = blockIdx.y * 32 + ty * 4 + i;
        int c = blockIdx.x * 32 + tx;
        tile[ty * 4 + i][tx] = W[(size_t)d * CC + c];
    }
    __syncthreads();
    #pragma unroll
    for (int i = 0; i < 4; ++i) {
        int c = blockIdx.x * 32 + ty * 4 + i;
        int d = blockIdx.y * 32 + tx;
        Wt[(size_t)c * CC + d] = to_bf16(tile[tx][ty * 4 + i]);
    }
}

// ------------------------ embedding + time-mix -----------------------------
__global__ __launch_bounds__(256) void embed_mix(
    const int*   __restrict__ tok,   const float* __restrict__ emb,
    const float* __restrict__ mk,    const float* __restrict__ mv,
    const float* __restrict__ mr,    const float* __restrict__ xx_init,
    __bf16* __restrict__ xk, __bf16* __restrict__ xv, __bf16* __restrict__ xr) {
    const int idx = blockIdx.x * blockDim.x + threadIdx.x;   // over B*T*C
    const int c  = idx & (CC - 1);
    const int bt = idx >> 10;
    const int t  = bt & (TT - 1);
    const int b  = bt >> 10;
    const float x  = emb[(size_t)tok[bt] * CC + c];
    const float xp = (t == 0) ? xx_init[b * CC + c]
                              : emb[(size_t)tok[bt - 1] * CC + c];
    const float k = mk[c], v = mv[c], r = mr[c];
    xk[idx] = to_bf16(x * k + xp * (1.0f - k));
    xv[idx] = to_bf16(x * v + xp * (1.0f - v));
    xr[idx] = to_bf16(x * r + xp * (1.0f - r));
}

// --------------------------- WMMA bf16 GEMM --------------------------------
// D[M,N] = A[M,K] * Bt[K,N], M=8192, N=K=1024.
// Block: 256 thr (8 waves: 4 M-waves x 2 N-groups). Block tile 128x128.
// Wave tile 32x64 -> 8 accumulators; LDS double-buffers A(128x32) + B(32x128).
__global__ __launch_bounds__(256) void gemm_bf16_wmma(
    const __bf16* __restrict__ A, const __bf16* __restrict__ Bt,
    float* __restrict__ D) {
    __shared__ __bf16 ldsA[2][128][40];    // 32 halves/row + 8 pad (16B aligned)
    __shared__ __bf16 ldsB[2][32][136];    // 128 halves/row + 8 pad

    const int tid  = threadIdx.x;
    const int lane = tid & 31;
    const int wave = tid >> 5;
    const int half = lane >> 4;            // 0: lanes 0-15, 1: lanes 16-31
    const int l    = lane & 15;
    const int wm   = wave >> 1;            // 0..3  (M position)
    const int wn   = wave & 1;             // 0..1  (N position)
    const int m0   = blockIdx.y * 128;
    const int n0   = blockIdx.x * 128;

    // cooperative staging: 512 16B chunks per tile, 2 per thread
    const int arow0 = tid >> 2,        acol0 = (tid & 3) * 8;   // A: 4 chunks/row
    const int arow1 = 64 + (tid >> 2);
    const int brow0 = tid >> 4,        bcol0 = (tid & 15) * 8;  // B: 16 chunks/row
    const int brow1 = 16 + (tid >> 4);

    v8f acc[2][4];
    #pragma unroll
    for (int mi = 0; mi < 2; ++mi)
        #pragma unroll
        for (int nj = 0; nj < 4; ++nj) acc[mi][nj] = (v8f){};

    v8bf ra0, ra1, rb0, rb1;
    // prologue: stage k-tile 0
    ra0 = *(const v8bf*)(A  + (size_t)(m0 + arow0) * CC + acol0);
    ra1 = *(const v8bf*)(A  + (size_t)(m0 + arow1) * CC + acol0);
    rb0 = *(const v8bf*)(Bt + (size_t)brow0 * CC + n0 + bcol0);
    rb1 = *(const v8bf*)(Bt + (size_t)brow1 * CC + n0 + bcol0);
    *(v8bf*)&ldsA[0][arow0][acol0] = ra0;
    *(v8bf*)&ldsA[0][arow1][acol0] = ra1;
    *(v8bf*)&ldsB[0][brow0][bcol0] = rb0;
    *(v8bf*)&ldsB[0][brow1][bcol0] = rb1;
    __syncthreads();

    #pragma unroll            // full unroll: LDS offsets become immediates,
    for (int kk = 0; kk < CC; kk += 32) {   // waits batch across iterations
        const int  buf     = (kk >> 5) & 1;
        const bool hasNext = (kk + 32) < CC;
        if (hasNext) {   // issue next-tile global loads early (overlap)
            ra0 = *(const v8bf*)(A  + (size_t)(m0 + arow0) * CC + (kk + 32) + acol0);
            ra1 = *(const v8bf*)(A  + (size_t)(m0 + arow1) * CC + (kk + 32) + acol0);
            rb0 = *(const v8bf*)(Bt + (size_t)(kk + 32 + brow0) * CC + n0 + bcol0);
            rb1 = *(const v8bf*)(Bt + (size_t)(kk + 32 + brow1) * CC + n0 + bcol0);
        }

        // Preload ALL fragments for this K-step (A: 4 x 16B loads, B: 64 x 16-bit loads), then issue
        // the 8 WMMAs back-to-back so DS waits are batched, not per-WMMA.
        // A fragments (16x32): lanes 0-15 -> K {0..7,16..23}; 16-31 -> {8..15,24..31}
        v16bf afrag[2];
        #pragma unroll
        for (int mi = 0; mi < 2; ++mi) {
            const __bf16* ap = &ldsA[buf][wm * 32 + mi * 16 + l][0];
            *(v8bf*)&afrag[mi]       = *(const v8bf*)(ap + 8 * half);
            *((v8bf*)&afrag[mi] + 1) = *(const v8bf*)(ap + 16 + 8 * half);
        }
        // B fragments (32x16): lane holds column N=l; lanes 0-15 -> K {0..7,16..23}; 16-31 -> {8..15,24..31}
        v16bf bfrag[4];
        #pragma unroll
        for (int nj = 0; nj < 4; ++nj) {
            const __bf16* bp = &ldsB[buf][0][wn * 64 + nj * 16 + l];
            #pragma unroll
            for (int s = 0; s < 16; ++s)
                bfrag[nj][s] = bp[((s < 8) ? (8 * half + s) : (16 + 8 * half + (s - 8))) * 136];
        }
        #pragma unroll
        for (int nj = 0; nj < 4; ++nj) {
            acc[0][nj] = __builtin_amdgcn_wmma_f32_16x16x32_bf16(
                false, afrag[0], false, bfrag[nj], (short)0, acc[0][nj], false, false);
            acc[1][nj] = __builtin_amdgcn_wmma_f32_16x16x32_bf16(
                false, afrag[1], false, bfrag[nj], (short)0, acc[1][nj], false, false);
        }

        if (hasNext) {   // fill the other buffer (consumed last iter, safe)
            *(v8bf*)&ldsA[buf ^ 1][arow0][acol0] = ra0;
            *(v8bf*)&ldsA[buf ^ 1][arow1][acol0] = ra1;
            *(v8bf*)&ldsB[buf ^ 1][brow0][bcol0] = rb0;
            *(v8bf*)&ldsB[buf ^ 1][brow1][bcol0] = rb1;
        }
        __syncthreads();
    }

    // C/D layout: VGPR i, lanes 0-15 -> (M=+i, N=l); lanes 16-31 -> (M=+8+i, N=l)
    #pragma unroll
    for (int mi = 0; mi < 2; ++mi) {
        float* Drow = D + (size_t)(m0 + wm * 32 + mi * 16 + 8 * half) * CC
                        + n0 + wn * 64 + l;
        #pragma unroll
        for (int i = 0; i < 8; ++i) {
            float* dp = Drow + (size_t)i * CC;
            dp[0]  = acc[mi][0][i];
            dp[16] = acc[mi][1][i];
            dp[32] = acc[mi][2][i];
            dp[48] = acc[mi][3][i];
        }
    }
}

// ------------------- WKV scan: chunk-parallel (3 phases) -------------------
// Stabilized state (p,a,b) represents A = a*e^p, B = b*e^p.
// Step (reference):  pn = max(p+w, k); a' = e^{p+w-pn}a + e^{k-pn}v; ...
// phase1: per-(b,chunk,c) local prefix states (identity start), stored per t.
__global__ __launch_bounds__(256) void wkv_phase1(
    const float* __restrict__ Kf, const float* __restrict__ Vf,
    const float* __restrict__ td,
    float* __restrict__ locp, float* __restrict__ loca, float* __restrict__ locb,
    float* __restrict__ chp,  float* __restrict__ cha,  float* __restrict__ chb) {
    const int idx = blockIdx.x * blockDim.x + threadIdx.x;   // B*NCH*C
    const int c  = idx & (CC - 1);
    const int bj = idx >> 10;          // b*NCH + j
    const int j  = bj & (NCH - 1);
    const int b  = bj >> 4;
    const float w = -__expf(td[c]);
    float p = -1e38f, a = 0.0f, bs = 0.0f;
    const size_t base = (size_t)b * TT * CC + (size_t)j * CL * CC + c;
    for (int t = 0; t < CL; ++t) {
        const size_t off = base + (size_t)t * CC;
        locp[off] = p; loca[off] = a; locb[off] = bs;        // incoming local state
        const float kt = Kf[off], vt = Vf[off];
        const float ww = p + w;
        const float pn = fmaxf(ww, kt);
        const float e1 = __expf(ww - pn), e2 = __expf(kt - pn);
        a = e1 * a + e2 * vt; bs = e1 * bs + e2; p = pn;
    }
    const size_t coff = (size_t)bj * CC + c;
    chp[coff] = p; cha[coff] = a; chb[coff] = bs;
}

// phase2: sequential combine over 16 chunks per channel; emits chunk-start
// global states and the final hx = (aa+bb)/2 in the reference representation.
__global__ __launch_bounds__(256) void wkv_phase2(
    const float* __restrict__ chp, const float* __restrict__ cha,
    const float* __restrict__ chb, const float* __restrict__ td,
    const float* __restrict__ aa0, const float* __restrict__ bb0,
    const float* __restrict__ pp0,
    float* __restrict__ stp, float* __restrict__ sta, float* __restrict__ stb,
    float* __restrict__ hx) {
    const int idx = blockIdx.x * blockDim.x + threadIdx.x;   // B*C
    const int c = idx & (CC - 1);
    const int b = idx >> 10;
    const float w  = -__expf(td[c]);
    const float wL = w * (float)CL;
    float p = pp0[idx], a = aa0[idx], bs = bb0[idx];
    for (int j = 0; j < NCH; ++j) {
        const size_t soff = ((size_t)b * NCH + j) * CC + c;
        stp[soff] = p; sta[soff] = a; stb[soff] = bs;
        const float cp = chp[soff], ca = cha[soff], cb = chb[soff];
        const float ps = p + wL;
        const float pn = fmaxf(ps, cp);
        const float e1 = __expf(ps - pn), e2 = __expf(cp - pn);
        a = e1 * a + e2 * ca; bs = e1 * bs + e2 * cb; p = pn;
    }
    hx[idx] = (a + bs) * 0.5f;
}

// phase3: fully parallel output: incoming = e^{w*tau}*chunkStart (+) local,
// then the reference output formula + fused sigmoid(r)*wkv -> bf16.
__global__ __launch_bounds__(256) void wkv_phase3(
    const float* __restrict__ Kf, const float* __restrict__ Vf,
    const float* __restrict__ Rp,
    const float* __restrict__ locp, const float* __restrict__ loca,
    const float* __restrict__ locb,
    const float* __restrict__ stp,  const float* __restrict__ sta,
    const float* __restrict__ stb,
    const float* __restrict__ td,   const float* __restrict__ tf,
    __bf16* __restrict__ wkvr) {
    const int idx = blockIdx.x * blockDim.x + threadIdx.x;   // B*T*C
    const int c  = idx & (CC - 1);
    const int bt = idx >> 10;
    const int t  = bt & (TT - 1);
    const int b  = bt >> 10;
    const int j   = t >> 6;            // CL = 64
    const int tau = t & (CL - 1);
    const float w = -__expf(td[c]);
    const float u = tf[c];
    const size_t soff = ((size_t)b * NCH + j) * CC + c;
    const float ps  = stp[soff] + w * (float)tau;
    const float pl  = locp[idx];
    const float pin = fmaxf(ps, pl);
    const float e1  = __expf(ps - pin), e2 = __expf(pl - pin);
    const float aa  = e1 * sta[soff] + e2 * loca[idx];
    const float bb  = e1 * stb[soff] + e2 * locb[idx];
    const float kt = Kf[idx], vt = Vf[idx];
    const float ww = u + kt;
    const float p  = fmaxf(pin, ww);
    const float ea = __expf(pin - p), eb = __expf(ww - p);
    const float out = (ea * aa + eb * vt) / (ea * bb + eb);
    const float r = 1.0f / (1.0f + __expf(-Rp[idx]));
    wkvr[idx] = to_bf16(r * out);
}

// ------------------ WKV scan: serial fallback (small ws) -------------------
__global__ __launch_bounds__(256) void wkv_scan_serial(
    const float* __restrict__ Kf, const float* __restrict__ Vf,
    const float* __restrict__ Rp,
    const float* __restrict__ td, const float* __restrict__ tf,
    const float* __restrict__ aa_init, const float* __restrict__ bb_init,
    const float* __restrict__ pp_init,
    __bf16* __restrict__ wkvr, float* __restrict__ hx) {
    const int idx = blockIdx.x * blockDim.x + threadIdx.x;   // B*C
    const int c = idx & (CC - 1);
    const int b = idx >> 10;
    const float u = tf[c];
    const float w = -__expf(td[c]);
    float aa = aa_init[idx], bb = bb_init[idx], pp = pp_init[idx];
    const size_t base = (size_t)b * TT * CC + c;
    for (int t = 0; t < TT; ++t) {
        const size_t off = base + (size_t)t * CC;
        const float kt = Kf[off], vt = Vf[off];
        float ww = u + kt;
        float p  = fmaxf(pp, ww);
        float e1 = __expf(pp - p), e2 = __expf(ww - p);
        const float out = (e1 * aa + e2 * vt) / (e1 * bb + e2);
        ww = pp + w;
        p  = fmaxf(ww, kt);
        e1 = __expf(ww - p); e2 = __expf(kt - p);
        aa = e1 * aa + e2 * vt; bb = e1 * bb + e2; pp = p;
        const float r = 1.0f / (1.0f + __expf(-Rp[off]));
        wkvr[off] = to_bf16(r * out);
    }
    hx[idx] = (aa + bb) * 0.5f;
}

// ------------------------------- launcher ----------------------------------
extern "C" void kernel_launch(void* const* d_in, const int* in_sizes, int n_in,
                              void* d_out, int out_size, void* d_ws, size_t ws_size,
                              hipStream_t stream) {
    const int*   tok = (const int*)  d_in[0];
    const float* emb = (const float*)d_in[1];
    const float* Wk  = (const float*)d_in[2];
    const float* Wv  = (const float*)d_in[3];
    const float* Wr  = (const float*)d_in[4];
    const float* Wo  = (const float*)d_in[5];
    const float* td  = (const float*)d_in[6];
    const float* tf  = (const float*)d_in[7];
    const float* mk  = (const float*)d_in[8];
    const float* mv  = (const float*)d_in[9];
    const float* mr  = (const float*)d_in[10];
    const float* xx0 = (const float*)d_in[11];
    const float* aa0 = (const float*)d_in[12];
    const float* bb0 = (const float*)d_in[13];
    const float* pp0 = (const float*)d_in[14];

    float* out_rwkv = (float*)d_out;
    float* out_hx   = out_rwkv + (size_t)BT * CC;

    // workspace layout
    char*  ws  = (char*)d_ws;
    size_t off = 0;
    const size_t wbytes = (size_t)CC * CC * sizeof(__bf16);        // 2 MB
    const size_t abytes = (size_t)BT * CC * sizeof(__bf16);        // 16 MB
    const size_t fbytes = (size_t)BT * CC * sizeof(float);         // 32 MB
    const size_t cbytes = (size_t)BB * NCH * CC * sizeof(float);   // 512 KB
    __bf16* WkT  = (__bf16*)(ws + off); off += wbytes;
    __bf16* WvT  = (__bf16*)(ws + off); off += wbytes;
    __bf16* WrT  = (__bf16*)(ws + off); off += wbytes;
    __bf16* WoT  = (__bf16*)(ws + off); off += wbytes;
    __bf16* xk   = (__bf16*)(ws + off); off += abytes;
    __bf16* xv   = (__bf16*)(ws + off); off += abytes;
    __bf16* xr   = (__bf16*)(ws + off); off += abytes;
    float*  Kf   = (float*) (ws + off); off += fbytes;
    float*  Vf   = (float*) (ws + off); off += fbytes;
    float*  Rp   = (float*) (ws + off); off += fbytes;
    __bf16* wkvr = (__bf16*)(ws + off); off += abytes;             // 168 MB
    float*  locp = (float*) (ws + off); off += fbytes;
    float*  loca = (float*) (ws + off); off += fbytes;
    float*  locb = (float*) (ws + off); off += fbytes;
    float*  chp  = (float*) (ws + off); off += cbytes;
    float*  cha  = (float*) (ws + off); off += cbytes;
    float*  chb  = (float*) (ws + off); off += cbytes;
    float*  stp  = (float*) (ws + off); off += cbytes;
    float*  sta  = (float*) (ws + off); off += cbytes;
    float*  stb  = (float*) (ws + off); off += cbytes;             // ~267 MB
    const bool chunked = (ws_size >= off);

    // 1) convert+transpose weights to bf16 [C, D]
    {
        dim3 tb(32, 8), tg(CC / 32, CC / 32);
        w_transpose_bf16<<<tg, tb, 0, stream>>>(Wk, WkT);
        w_transpose_bf16<<<tg, tb, 0, stream>>>(Wv, WvT);
        w_transpose_bf16<<<tg, tb, 0, stream>>>(Wr, WrT);
        w_transpose_bf16<<<tg, tb, 0, stream>>>(Wo, WoT);
    }

    // 2) embedding gather + time-mix -> bf16 activations
    embed_mix<<<(BT * CC) / 256, 256, 0, stream>>>(tok, emb, mk, mv, mr, xx0,
                                                   xk, xv, xr);

    // 3) K / V / R-preact GEMMs (WMMA bf16 -> f32)
    {
        dim3 g(CC / 128, BT / 128);
        gemm_bf16_wmma<<<g, 256, 0, stream>>>(xk, WkT, Kf);
        gemm_bf16_wmma<<<g, 256, 0, stream>>>(xv, WvT, Vf);
        gemm_bf16_wmma<<<g, 256, 0, stream>>>(xr, WrT, Rp);
    }

    // 4) WKV recurrence + fused sigmoid(r)*wkv, hx output
    if (chunked) {
        wkv_phase1<<<(BB * NCH * CC) / 256, 256, 0, stream>>>(
            Kf, Vf, td, locp, loca, locb, chp, cha, chb);
        wkv_phase2<<<(BB * CC) / 256, 256, 0, stream>>>(
            chp, cha, chb, td, aa0, bb0, pp0, stp, sta, stb, out_hx);
        wkv_phase3<<<(BT * CC) / 256, 256, 0, stream>>>(
            Kf, Vf, Rp, locp, loca, locb, stp, sta, stb, td, tf, wkvr);
    } else {
        wkv_scan_serial<<<(BB * CC) / 256, 256, 0, stream>>>(
            Kf, Vf, Rp, td, tf, aa0, bb0, pp0, wkvr, out_hx);
    }

    // 5) output projection GEMM straight into d_out
    {
        dim3 g(CC / 128, BT / 128);
        gemm_bf16_wmma<<<g, 256, 0, stream>>>(wkvr, WoT, out_rwkv);
    }
}
